// gfmm_32435593020276
// MI455X (gfx1250) — matrix operand order repaired and checked by reading, not compiled, not run on hardware
//
#include <hip/hip_runtime.h>
#include <hip/hip_bf16.h>
#include <math.h>

// MI455X / gfx1250, wave32. bf16 WMMA (v_wmma_f32_16x16x32_bf16), f32 accum.
// Memory-bound: read x once, write out once (~256MiB -> ~11us @ 23.3TB/s).
// 32 tokens/block (2 M-tiles) to amortize packed-Wp L2 panel traffic 2x vs
// a 16-token tile; B panels reused across both M-tiles per load.

typedef __attribute__((ext_vector_type(16))) __bf16 v16bf;
typedef __attribute__((ext_vector_type(8)))  float  v8f;

#define DIMC    256
#define HEADS   8
#define HEAD_D  32
#define NP      64
#define HWTOK   16384        // H*W = 128*128
#define SCALE_F 0.17677669529663687f  // 32^-0.5

#if defined(__has_builtin)
# if __has_builtin(__builtin_amdgcn_global_load_async_to_lds_b32)
#  define USE_ASYNC_LDS 1
# endif
#endif

#ifdef USE_ASYNC_LDS
typedef __attribute__((address_space(1))) int g_int;   // "__device__ int *"
typedef __attribute__((address_space(3))) int l_int;   // "__shared__ int *"
#endif

// ---------------------------------------------------------------------------
// K0: 16x16 non-overlapping average pool.  grid = B*64, block = 256 (one ch)
// ---------------------------------------------------------------------------
__global__ __launch_bounds__(256)
void pool_kernel(const float* __restrict__ x, float* __restrict__ xp)
{
    const int blk = blockIdx.x;          // b*64 + m
    const int b = blk >> 6, m = blk & 63;
    const int c = threadIdx.x;
    const int mh = m >> 3, mw = m & 7;
    const float* p = x + ((size_t)(b * DIMC + c) * 128 + mh * 16) * 128 + mw * 16;
    float s = 0.f;
    #pragma unroll 4
    for (int i = 0; i < 16; ++i)
        for (int j = 0; j < 16; ++j) s += p[i * 128 + j];
    xp[(size_t)blk * DIMC + c] = s * (1.f / 256.f);
}

// ---------------------------------------------------------------------------
// K1: SR path per pooled token: Wsr GEMV + LN + exact GELU, then k and v.
//     v written straight into WMMA B panels (bf16, 16x16x32), contiguous k order used here:
//     lanes 0-15 hold K=0..15 (e=K), lanes 16-31 hold K=16..31 (e=K-16).
// ---------------------------------------------------------------------------
__global__ __launch_bounds__(256)
void sr_kernel(const float* __restrict__ xp,  const float* __restrict__ Wsr,
               const float* __restrict__ bsr, const float* __restrict__ gamma,
               const float* __restrict__ beta, const float* __restrict__ Wk,
               const float* __restrict__ Wv,  float* __restrict__ kws,
               __bf16* __restrict__ vpack)
{
    __shared__ float xs[256], xt[256], r1[256], r2[256];
    const int blk = blockIdx.x;          // b*64 + m
    const int b = blk >> 6, m = blk & 63;
    const int o = threadIdx.x;

    xs[o] = xp[(size_t)blk * DIMC + o];
    __syncthreads();

    float acc = bsr[o];
    #pragma unroll 4
    for (int c = 0; c < DIMC; ++c) acc = fmaf(xs[c], Wsr[o * DIMC + c], acc);

    r1[o] = acc; r2[o] = acc * acc;
    __syncthreads();
    for (int s = 128; s > 0; s >>= 1) {
        if (o < s) { r1[o] += r1[o + s]; r2[o] += r2[o + s]; }
        __syncthreads();
    }
    const float mu  = r1[0] * (1.f / 256.f);
    const float var = r2[0] * (1.f / 256.f) - mu * mu;
    const float tn  = (acc - mu) * rsqrtf(var + 1e-5f) * gamma[o] + beta[o];
    const float g   = 0.5f * tn * (1.f + erff(tn * 0.70710678118654752f)); // exact GELU
    xt[o] = g;
    __syncthreads();

    float vv = 0.f;
    #pragma unroll 4
    for (int c = 0; c < DIMC; ++c) vv = fmaf(xt[c], Wv[o * DIMC + c], vv);

    // scatter into packed B panels: [b][h][kb][nt][lane][e]
    const int h  = o >> 5, d = o & 31;
    const int kb = m >> 5, mm = m & 31;
    const int lh = mm >> 4, e = mm & 15;
    const int ln = lh * 16 + (d & 15);
    const int nt = (d >> 4) & 1;
    vpack[((size_t)((((b * 8 + h) * 2 + kb) * 2 + nt) * 32 + ln) << 4) + e] = (__bf16)vv;

    if (o < HEADS) {
        float kk = 0.f;
        #pragma unroll 4
        for (int c = 0; c < DIMC; ++c) kk = fmaf(xt[c], Wk[o * DIMC + c], kk);
        kws[(b * 8 + o) * 64 + m] = kk;
    }
}

// ---------------------------------------------------------------------------
// K2: pre-pack Wp^T into bf16 WMMA B panels: [ot(16)][kb(8)][lane(32)][e(16)]
// ---------------------------------------------------------------------------
__global__ __launch_bounds__(256)
void wp_pack_kernel(const float* __restrict__ Wp, __bf16* __restrict__ wppack)
{
    const int id = blockIdx.x * 256 + threadIdx.x;   // 0..65535
    const int e  = id & 15;
    const int ln = (id >> 4) & 31;
    const int kb = (id >> 9) & 7;
    const int ot = id >> 12;
    const int k  = kb * 32 + ((ln >> 4) << 4) + e;
    const int o  = ot * 16 + (ln & 15);
    wppack[id] = (__bf16)Wp[o * DIMC + k];
}

// ---------------------------------------------------------------------------
// K3: main pass. 32 tokens/block, 256 threads = 8 wave32 (one head per wave).
//  1) prefetch packed panels (global_prefetch_b8); async-stage Wq/k into LDS
//  2) per (tok,head)=thread: q = <x,Wq[h]> (direct global, 128B/wave
//     coalesced), 64-wide softmax vs k -> attn bf16 LDS (A-operand friendly)
//  3) attention: per wave 1 head, 2 M-tiles share each B panel, WMMA bf16
//  4) projection: per wave 2 o-tiles x 8 K-chunks, B panel reused across both
//     M-tiles; bias; out-tile overlays dead attn region
//  5) 128B-contiguous stores to out[B,C,H,W]
// LDS map (64256 B): attn[8][32][72]bf16 (36864, union out[256][33]f32 33792)
//                    | wq[8][260]f32 8320 | k[8][68]f32 2176 | y[32][264]bf16 16896
// ---------------------------------------------------------------------------
__global__ __launch_bounds__(256)
void attn_proj_kernel(const float*  __restrict__ x,
                      const float*  __restrict__ Wq,
                      const float*  __restrict__ kws,
                      const __bf16* __restrict__ vpack,
                      const __bf16* __restrict__ wppack,
                      const float*  __restrict__ bp,
                      float* __restrict__ out)
{
    __shared__ __align__(16) char smem[64256];
    __bf16* s_attn = (__bf16*)smem;                        // [8][32][72]
    float*  s_os   = (float*)smem;                         // [256][33] (attn dead)
    float*  s_wq   = (float*)(smem + 36864);               // [8][260]
    float*  s_k    = (float*)(smem + 36864 + 8320);        // [8][68]
    __bf16* s_y    = (__bf16*)(smem + 36864 + 8320 + 2176);// [32][264]

    const int b    = blockIdx.x >> 9;          // 512 tiles of 32 tokens / batch
    const int n0   = (blockIdx.x & 511) << 5;
    const int t    = threadIdx.x;
    const int lane = t & 31;
    const int wave = t >> 5;

    // ---- warm L0/L2 with the panels this block streams (global_prefetch_b8)
    {
        const __bf16* vp = vpack + (size_t)b * 8192;       // this batch: 16KB
        if (t < 128) __builtin_prefetch(vp + t * 64, 0, 3);
        #pragma unroll
        for (int i = 0; i < 4; ++i)                        // wppack: 128KB
            __builtin_prefetch(wppack + ((size_t)t * 4 + i) * 64, 0, 3);
    }

    // ---- stage Wq and k into LDS (async DMA path if available) ----
#ifdef USE_ASYNC_LDS
    {
        #pragma unroll
        for (int i = 0; i < 8; ++i) {                      // Wq: 2048 dwords
            const int id = t + i * 256;
            const int h = id >> 8, c = id & 255;
            __builtin_amdgcn_global_load_async_to_lds_b32(
                (g_int*)(Wq + h * DIMC + c),
                (l_int*)(s_wq + h * 260 + c), 0, 0);
        }
        #pragma unroll
        for (int i = 0; i < 2; ++i) {                      // k: 512 dwords
            const int id = t + i * 256;
            const int h = id >> 6, m = id & 63;
            __builtin_amdgcn_global_load_async_to_lds_b32(
                (g_int*)(kws + (b * 8 + h) * 64 + m),
                (l_int*)(s_k + h * 68 + m), 0, 0);
        }
        asm volatile("s_wait_asynccnt 0x0" ::: "memory");
    }
#else
    for (int idx = t; idx < 2048; idx += 256) {
        const int h = idx >> 8, c = idx & 255;
        s_wq[h * 260 + c] = Wq[h * DIMC + c];
    }
    for (int idx = t; idx < 512; idx += 256) {
        const int h = idx >> 6, m = idx & 63;
        s_k[h * 68 + m] = kws[(b * 8 + h) * 64 + m];
    }
#endif
    __syncthreads();

    // ---- q + rank-1 softmax, one (tok,head) per thread ----
    {
        const int tok = t & 31, h = t >> 5;
        const float* xcol = x + (size_t)b * DIMC * HWTOK + n0 + tok;
        float q0 = 0.f, q1 = 0.f, q2 = 0.f, q3 = 0.f;
        #pragma unroll 4
        for (int c = 0; c < DIMC; c += 4) {                // 4-way ILP
            q0 = fmaf(xcol[(size_t)(c + 0) * HWTOK], s_wq[h * 260 + c + 0], q0);
            q1 = fmaf(xcol[(size_t)(c + 1) * HWTOK], s_wq[h * 260 + c + 1], q1);
            q2 = fmaf(xcol[(size_t)(c + 2) * HWTOK], s_wq[h * 260 + c + 2], q2);
            q3 = fmaf(xcol[(size_t)(c + 3) * HWTOK], s_wq[h * 260 + c + 3], q3);
        }
        const float q = ((q0 + q1) + (q2 + q3)) * SCALE_F;
        float mx = -3.4e38f;
        for (int m = 0; m < NP; ++m) mx = fmaxf(mx, q * s_k[h * 68 + m]);
        float sum = 0.f;
        for (int m = 0; m < NP; ++m) sum += __expf(q * s_k[h * 68 + m] - mx);
        const float inv = 1.f / sum;
        __bf16* ap = &s_attn[(h * 32 + tok) * 72];
        for (int m = 0; m < NP; ++m)
            ap[m] = (__bf16)(__expf(q * s_k[h * 68 + m] - mx) * inv);
    }
    __syncthreads();

    // ---- attention WMMAs: wave = head; 2 M-tiles share each B panel ----
    // A operand, k order used here (matches B packs): lanes0-15 K in {0..15}; lanes16-31 {16..31}
    const int row  = lane & 15;
    const int ksel = (lane & 16) ? 16 : 0;
    const int rowb = (lane >> 4) << 3;       // D: lanes16-31 hold M=8..15
    {
        const int h = wave;
        for (int nt = 0; nt < 2; ++nt) {
            v8f acc0 = {}, acc1 = {};
            #pragma unroll
            for (int kb = 0; kb < 2; ++kb) {
                const v16bf bv = *(const v16bf*)(vpack +
                    ((size_t)((((b * 8 + h) * 2 + kb) * 2 + nt) * 32 + lane) << 4));
                v16bf a0, a1;
                const __bf16* ap0 = &s_attn[(h * 32 + row)      * 72 + kb * 32 + ksel];
                const __bf16* ap1 = &s_attn[(h * 32 + 16 + row) * 72 + kb * 32 + ksel];
                #pragma unroll
                for (int e = 0; e < 8; ++e) {
                    a0[e] = ap0[e]; a0[e + 8] = ap0[e + 8];
                    a1[e] = ap1[e]; a1[e + 8] = ap1[e + 8];
                }
                acc0 = __builtin_amdgcn_wmma_f32_16x16x32_bf16(
                    false, a0, false, bv, (short)0, acc0, false, false);
                acc1 = __builtin_amdgcn_wmma_f32_16x16x32_bf16(
                    false, a1, false, bv, (short)0, acc1, false, false);
            }
            const int col = h * HEAD_D + nt * 16 + (lane & 15);
            #pragma unroll
            for (int j = 0; j < 8; ++j) {
                s_y[(rowb + j) * 264 + col]      = (__bf16)acc0[j];
                s_y[(16 + rowb + j) * 264 + col] = (__bf16)acc1[j];
            }
        }
    }
    __syncthreads();

    // ---- projection WMMAs: wave handles 2 o-tiles; B reused across M-tiles ----
    for (int oi = 0; oi < 2; ++oi) {
        const int ot = wave * 2 + oi;
        v8f acc0 = {}, acc1 = {};
        #pragma unroll
        for (int kb = 0; kb < 8; ++kb) {
            const v16bf bv = *(const v16bf*)(wppack +
                ((size_t)((ot * 8 + kb) * 32 + lane) << 4));
            v16bf a0, a1;
            const __bf16* ap0 = &s_y[row * 264 + kb * 32 + ksel];
            const __bf16* ap1 = &s_y[(16 + row) * 264 + kb * 32 + ksel];
            #pragma unroll
            for (int e = 0; e < 8; ++e) {
                a0[e] = ap0[e]; a0[e + 8] = ap0[e + 8];
                a1[e] = ap1[e]; a1[e + 8] = ap1[e + 8];
            }
            acc0 = __builtin_amdgcn_wmma_f32_16x16x32_bf16(
                false, a0, false, bv, (short)0, acc0, false, false);
            acc1 = __builtin_amdgcn_wmma_f32_16x16x32_bf16(
                false, a1, false, bv, (short)0, acc1, false, false);
        }
        const int o    = ot * 16 + (lane & 15);
        const float bb = bp[o];
        #pragma unroll
        for (int j = 0; j < 8; ++j) {
            s_os[o * 33 + rowb + j]      = acc0[j] + bb;   // attn region is dead; reuse
            s_os[o * 33 + 16 + rowb + j] = acc1[j] + bb;
        }
    }
    __syncthreads();

    // ---- coalesced store: 32 contiguous tokens (128B) per channel row ----
    for (int idx = t; idx < 8192; idx += 256) {
        const int o = idx >> 5, tk = idx & 31;
        out[(size_t)(b * DIMC + o) * HWTOK + n0 + tk] = s_os[o * 33 + tk];
    }
}

// ---------------------------------------------------------------------------
extern "C" void kernel_launch(void* const* d_in, const int* in_sizes, int n_in,
                              void* d_out, int out_size, void* d_ws, size_t ws_size,
                              hipStream_t stream)
{
    const float* x     = (const float*)d_in[0];
    const float* Wq    = (const float*)d_in[1];
    const float* Wk    = (const float*)d_in[2];
    const float* Wv    = (const float*)d_in[3];
    const float* Wsr   = (const float*)d_in[4];
    const float* bsr   = (const float*)d_in[5];
    const float* gamma = (const float*)d_in[6];
    const float* beta  = (const float*)d_in[7];
    const float* Wp    = (const float*)d_in[8];
    const float* bp    = (const float*)d_in[9];
    float* out = (float*)d_out;

    char* ws = (char*)d_ws;
    float*  xp    = (float*)(ws);              // 8*64*256*4      = 512 KiB
    float*  kws   = (float*)(ws + 524288);     // 8*8*64*4        =  16 KiB
    __bf16* vpack = (__bf16*)(ws + 540672);    // 8*8*2*2*32*16*2 = 256 KiB
    __bf16* wppk  = (__bf16*)(ws + 802816);    // 65536*2         = 128 KiB

    pool_kernel     <<<512,  256, 0, stream>>>(x, xp);
    sr_kernel       <<<512,  256, 0, stream>>>(xp, Wsr, bsr, gamma, beta, Wk, Wv, kws, vpack);
    wp_pack_kernel  <<<256,  256, 0, stream>>>(Wp, wppk);
    attn_proj_kernel<<<4096, 256, 0, stream>>>(x, Wq, kws, vpack, wppk, bp, out);
}
